// GLogicalLayer_73787538145663
// MI455X (gfx1250) — compile-verified, matrix operand order checked by reading, not run on hardware
//
#include <hip/hip_runtime.h>
#include <hip/hip_bf16.h>
#include <math.h>

typedef __bf16        v16bf __attribute__((ext_vector_type(16)));
typedef float         v8f   __attribute__((ext_vector_type(8)));
typedef unsigned int  v8u   __attribute__((ext_vector_type(8)));

#define B_   32
#define N_   4096
#define D_   128
#define K_   512
#define NPB_ 2048
#define M_   (B_*NPB_)   // 65536
#define E_   (B_*N_)     // 131072

static __device__ __forceinline__ unsigned short f2bf(float f) {
  unsigned int u = __float_as_uint(f);
  u += 0x7FFFu + ((u >> 16) & 1u);          // round-to-nearest-even
  return (unsigned short)(u >> 16);
}
static __device__ __forceinline__ float sigm(float x) {
  return 1.0f / (1.0f + expf(-x));
}
// K index inside a 16x32 bf16 A/B fragment for element pair v (ISA 7.12.2 layout)
static __device__ __forceinline__ int wk(int v, int kb) {
  return (v < 4) ? (kb + 2 * v) : (16 + kb + 2 * (v - 4));
}

// ---------------- kernel 0: init accumulators ----------------
__global__ void init_kernel(const float* __restrict__ tail_emd,
                            float* __restrict__ tail_agg,
                            float* __restrict__ new_hidden,
                            float* __restrict__ agg_att) {
  long long i = (long long)blockIdx.x * blockDim.x + threadIdx.x;
  if (i < (long long)M_ * D_) { tail_agg[i] = tail_emd[i]; new_hidden[i] = 0.f; }
  if (i < M_) agg_att[i] = 0.f;
}

// ---------------- kernel 1: query = [qh|qr|qt] @ W_q + b_q (32x384x128, tiny) ----------------
__global__ void query_kernel(const float* __restrict__ qh, const float* __restrict__ qr,
                             const float* __restrict__ qt, const float* __restrict__ Wq,
                             const float* __restrict__ bq, float* __restrict__ query) {
  int b = blockIdx.x, c = threadIdx.x;  // 128 threads
  float s = bq[c];
  for (int k = 0; k < D_; ++k) s += qh[b * D_ + k] * Wq[k * D_ + c];
  for (int k = 0; k < D_; ++k) s += qr[b * D_ + k] * Wq[(D_ + k) * D_ + c];
  for (int k = 0; k < D_; ++k) s += qt[b * D_ + k] * Wq[(2 * D_ + k) * D_ + c];
  query[b * D_ + c] = s;
}

// ---------------- kernel 2: fused main pass (WMMA GEMM + attention + segment sums) ----------------
__global__ __launch_bounds__(256) void fused_kernel(
    const float* __restrict__ q_head, const float* __restrict__ r_nb,
    const float* __restrict__ t_nb,   const float* __restrict__ tm_nb,
    const float* __restrict__ hidden, const int* __restrict__ tail_index,
    const float* __restrict__ W_a,    const float* __restrict__ b_a,
    const float* __restrict__ W_att,  const float* __restrict__ b_att,
    const float* __restrict__ W_rule, const float* __restrict__ b_rule,
    const float* __restrict__ query,
    float* __restrict__ tail_agg, float* __restrict__ new_hidden,
    float* __restrict__ agg_att) {
  __shared__ unsigned short Xs[64 * 130];    // 64 rows x 128 bf16, padded
  __shared__ unsigned short WsT[128 * 132];  // W_a chunk transposed, bf16, padded
  __shared__ float att1_l[64], att2_l[64], att_s[64];
  __shared__ int   tidx_s[64];
  __shared__ float qh_s[D_], q_s[D_], ba_s[D_], wr_s[D_];
  __shared__ float watt_s[3 * D_];

  const int tid = threadIdx.x;
  const int e0  = blockIdx.x * 64;   // 64 rows per block; never straddles a batch
  const int bb  = e0 >> 12;          // e0 / N_

  for (int i = tid; i < D_; i += 256) {
    qh_s[i] = q_head[bb * D_ + i];
    q_s[i]  = query[bb * D_ + i];
    ba_s[i] = b_a[i];
    wr_s[i] = W_rule[i];
  }
  for (int i = tid; i < 3 * D_; i += 256) watt_s[i] = W_att[i];
  if (tid < 64) { att1_l[tid] = 0.f; att2_l[tid] = 0.f; tidx_s[tid] = tail_index[e0 + tid]; }

  const int lane = tid & 31, wid = tid >> 5;
  const int rg = wid & 3, nh = wid >> 2;       // 4 row-groups x 2 col-halves
  const int r0 = rg * 16;
  const int mrow = lane & 15;
  const int kb = (lane < 16) ? 0 : 8;

  v8f acc[4] = {};

  for (int c = 0; c < 3; ++c) {  // K chunks: r / t / time (the 384-concat splits exactly)
    const float* src = (c == 0) ? r_nb : ((c == 1) ? t_nb : tm_nb);
    for (int i = tid; i < 64 * D_; i += 256) {
      int r = i >> 7, col = i & 127;
      Xs[r * 130 + col] = f2bf(src[(long long)(e0 + r) * D_ + col]);
    }
    for (int i = tid; i < D_ * D_; i += 256) {
      int k = i >> 7, n = i & 127;
      WsT[n * 132 + k] = f2bf(W_a[(c * D_ + k) * D_ + n]);
    }
    __syncthreads();
#pragma unroll
    for (int kt = 0; kt < 4; ++kt) {
      const int k0 = kt * 32;
      v8u au;
#pragma unroll
      for (int v = 0; v < 8; ++v)
        au[v] = *(const unsigned int*)&Xs[(r0 + mrow) * 130 + k0 + wk(v, kb)];
      v16bf a = __builtin_bit_cast(v16bf, au);
#pragma unroll
      for (int t = 0; t < 4; ++t) {
        int col = nh * 64 + t * 16 + mrow;
        v8u bu;
#pragma unroll
        for (int v = 0; v < 8; ++v)
          bu[v] = *(const unsigned int*)&WsT[col * 132 + k0 + wk(v, kb)];
        v16bf bmat = __builtin_bit_cast(v16bf, bu);
        acc[t] = __builtin_amdgcn_wmma_f32_16x16x32_bf16(
            false, a, false, bmat, (short)0, acc[t], false, false);
      }
    }
    __syncthreads();
  }

  // att1 logits from accumulator fragments: lane holds one column per tile, VGPR j = row
  float part[8];
#pragma unroll
  for (int j = 0; j < 8; ++j) part[j] = 0.f;
#pragma unroll
  for (int t = 0; t < 4; ++t) {
    int col = nh * 64 + t * 16 + mrow;
    float w0 = watt_s[col], w1 = watt_s[D_ + col], w2 = watt_s[2 * D_ + col];
    float q = q_s[col], ba = ba_s[col];
#pragma unroll
    for (int j = 0; j < 8; ++j) {
      float nv = acc[t][j] + ba;
      part[j] += q * nv * w0 + (q - nv) * w1 + (q + nv) * w2;
    }
  }
#pragma unroll
  for (int j = 0; j < 8; ++j) {
#pragma unroll
    for (int msk = 1; msk < 16; msk <<= 1) part[j] += __shfl_xor(part[j], msk, 16);
  }
  if (mrow == 0) {
    int rowb = r0 + ((lane >= 16) ? 8 : 0);
#pragma unroll
    for (int j = 0; j < 8; ++j) atomicAdd(&att1_l[rowb + j], part[j]);
  }

  // att2 logits + new_hidden segment sum (single pass over hidden rows)
  for (int i = tid; i < 64 * D_; i += 256) {
    int r = i >> 7, col = i & 127;
    float h = hidden[(long long)(e0 + r) * D_ + col];
    atomicAdd(&att2_l[r], h * wr_s[col]);
    atomicAdd(&new_hidden[(long long)tidx_s[r] * D_ + col], h);
  }
  __syncthreads();

  if (tid < 64) {
    float at = 0.5f * (sigm(att1_l[tid] + b_att[0]) + sigm(att2_l[tid] + b_rule[0]));
    att_s[tid] = at;
    atomicAdd(&agg_att[tidx_s[tid]], at);
  }
  __syncthreads();

  // msg = att * (q_head + r + time), segment-summed into tail_agg (L2-resident atomics)
  for (int i = tid; i < 64 * D_; i += 256) {
    int r = i >> 7, col = i & 127;
    float mval = qh_s[col] + r_nb[(long long)(e0 + r) * D_ + col]
                          + tm_nb[(long long)(e0 + r) * D_ + col];
    atomicAdd(&tail_agg[(long long)tidx_s[r] * D_ + col], att_s[r] * mval);
  }
}

// ---------------- kernel 3: per-batch bitonic top-512-of-2048 ----------------
__global__ __launch_bounds__(256) void topk_kernel(
    const float* __restrict__ agg_att, const int* __restrict__ tail_nodes,
    int* __restrict__ idx_ws, int* __restrict__ nodes_out) {
  __shared__ float sv[NPB_];
  __shared__ int   si[NPB_];
  const int tid = threadIdx.x, b = blockIdx.x;
  for (int i = tid; i < NPB_; i += 256) { sv[i] = agg_att[b * NPB_ + i]; si[i] = i; }
  __syncthreads();
  for (int k = 2; k <= NPB_; k <<= 1) {
    for (int j = k >> 1; j > 0; j >>= 1) {
      for (int i = tid; i < NPB_; i += 256) {
        int ixj = i ^ j;
        if (ixj > i) {
          float va = sv[i], vb = sv[ixj];
          int ia = si[i], ib = si[ixj];
          bool aLess = (va < vb) || (va == vb && ia > ib);  // desc value, asc index
          bool desc = ((i & k) == 0);
          if (desc ? aLess : !aLess) { sv[i] = vb; sv[ixj] = va; si[i] = ib; si[ixj] = ia; }
        }
      }
      __syncthreads();
    }
  }
  for (int i = tid; i < K_; i += 256) {
    int g = b * NPB_ + si[i];
    idx_ws[b * K_ + i] = g;
    nodes_out[(b * K_ + i) * 2 + 0] = tail_nodes[g * 3 + 1];
    nodes_out[(b * K_ + i) * 2 + 1] = tail_nodes[g * 3 + 2];
  }
}

// ---------------- kernel 4: out_emd = tail_agg[idx] @ W_out + b_out (WMMA), hid gather ----------------
__global__ __launch_bounds__(256) void out_kernel(
    const float* __restrict__ tail_agg, const float* __restrict__ new_hidden,
    const int* __restrict__ idx_ws, const float* __restrict__ W_out,
    const float* __restrict__ b_out, float* __restrict__ emd_out,
    float* __restrict__ hid_out) {
  __shared__ unsigned short Xs[64 * 130];
  __shared__ unsigned short WoT[128 * 132];
  __shared__ int gi_s[64];
  const int tid = threadIdx.x;
  const int row0 = blockIdx.x * 64;
  if (tid < 64) gi_s[tid] = idx_ws[row0 + tid];
  for (int i = tid; i < D_ * D_; i += 256) {
    int k = i >> 7, n = i & 127;
    WoT[n * 132 + k] = f2bf(W_out[k * D_ + n]);
  }
  __syncthreads();
  for (int i = tid; i < 64 * D_; i += 256) {
    int r = i >> 7, col = i & 127;
    Xs[r * 130 + col] = f2bf(tail_agg[(long long)gi_s[r] * D_ + col]);
  }
  __syncthreads();

  const int lane = tid & 31, wid = tid >> 5;
  const int rg = wid & 3, nh = wid >> 2;
  const int r0 = rg * 16;
  const int mrow = lane & 15;
  const int kb = (lane < 16) ? 0 : 8;

  v8f acc[4] = {};
#pragma unroll
  for (int kt = 0; kt < 4; ++kt) {
    const int k0 = kt * 32;
    v8u au;
#pragma unroll
    for (int v = 0; v < 8; ++v)
      au[v] = *(const unsigned int*)&Xs[(r0 + mrow) * 130 + k0 + wk(v, kb)];
    v16bf a = __builtin_bit_cast(v16bf, au);
#pragma unroll
    for (int t = 0; t < 4; ++t) {
      int col = nh * 64 + t * 16 + mrow;
      v8u bu;
#pragma unroll
      for (int v = 0; v < 8; ++v)
        bu[v] = *(const unsigned int*)&WoT[col * 132 + k0 + wk(v, kb)];
      v16bf bmat = __builtin_bit_cast(v16bf, bu);
      acc[t] = __builtin_amdgcn_wmma_f32_16x16x32_bf16(
          false, a, false, bmat, (short)0, acc[t], false, false);
    }
  }
#pragma unroll
  for (int t = 0; t < 4; ++t) {
    int col = nh * 64 + t * 16 + mrow;
    float bo = b_out[col];
#pragma unroll
    for (int j = 0; j < 8; ++j) {
      int row = row0 + r0 + j + ((lane >= 16) ? 8 : 0);
      emd_out[(long long)row * D_ + col] = acc[t][j] + bo;
    }
  }
  for (int i = tid; i < 64 * D_; i += 256) {
    int r = i >> 7, col = i & 127;
    hid_out[(long long)(row0 + r) * D_ + col] = new_hidden[(long long)gi_s[r] * D_ + col];
  }
}

extern "C" void kernel_launch(void* const* d_in, const int* in_sizes, int n_in,
                              void* d_out, int out_size, void* d_ws, size_t ws_size,
                              hipStream_t stream) {
  (void)in_sizes; (void)n_in; (void)out_size; (void)ws_size;
  const float* q_head = (const float*)d_in[0];
  const float* q_rel  = (const float*)d_in[1];
  const float* q_time = (const float*)d_in[2];
  const float* r_nb   = (const float*)d_in[3];
  const float* t_nb   = (const float*)d_in[4];
  const float* tm_nb  = (const float*)d_in[5];
  const float* hidden = (const float*)d_in[6];
  const float* tail_emd = (const float*)d_in[7];
  const int* tail_index = (const int*)d_in[8];
  const int* tail_nodes = (const int*)d_in[9];
  const float* W_q = (const float*)d_in[10];
  const float* b_q = (const float*)d_in[11];
  const float* W_a = (const float*)d_in[12];
  const float* b_a = (const float*)d_in[13];
  const float* W_att  = (const float*)d_in[14];
  const float* b_att  = (const float*)d_in[15];
  const float* W_rule = (const float*)d_in[16];
  const float* b_rule = (const float*)d_in[17];
  const float* W_out  = (const float*)d_in[18];
  const float* b_out  = (const float*)d_in[19];

  float* ws       = (float*)d_ws;
  float* query_ws = ws;                               // 4096 f32
  float* agg_att  = ws + 4096;                        // M f32
  float* tail_agg = agg_att + M_;                     // M*D f32
  float* new_hid  = tail_agg + (size_t)M_ * D_;       // M*D f32
  int*   idx_ws   = (int*)(new_hid + (size_t)M_ * D_); // B*K int

  int*   out_i   = (int*)d_out;                       // new_nodes (B,K,2) int32
  float* out_f   = (float*)d_out;
  float* emd_out = out_f + (size_t)B_ * K_ * 2;       // (B,K,D) f32
  float* hid_out = emd_out + (size_t)B_ * K_ * D_;    // (B,K,D) f32

  init_kernel<<<(M_ * D_) / 256, 256, 0, stream>>>(tail_emd, tail_agg, new_hid, agg_att);
  query_kernel<<<B_, D_, 0, stream>>>(q_head, q_rel, q_time, W_q, b_q, query_ws);
  fused_kernel<<<E_ / 64, 256, 0, stream>>>(q_head, r_nb, t_nb, tm_nb, hidden, tail_index,
                                            W_a, b_a, W_att, b_att, W_rule, b_rule,
                                            query_ws, tail_agg, new_hid, agg_att);
  topk_kernel<<<B_, 256, 0, stream>>>(agg_att, tail_nodes, idx_ws, out_i);
  out_kernel<<<(B_ * K_) / 64, 256, 0, stream>>>(tail_agg, new_hid, idx_ws, W_out, b_out,
                                                 emd_out, hid_out);
}
